// ModifiedDeFTTSEMemoryEfficient_41601053229493
// MI455X (gfx1250) — matrix operand order repaired and checked by reading, not compiled, not run on hardware
//
#include <hip/hip_runtime.h>
#include <hip/hip_bf16.h>

typedef _Float16 half_t;
typedef __attribute__((ext_vector_type(16))) _Float16 v16h;
typedef __attribute__((ext_vector_type(8)))  float    v8f;

#define CCn   96
#define TTn   192
#define FFn   256
#define TFP   (TTn*FFn)      // 49152
#define BQn   2
#define NPx   (BQn*TFP)      // 98304 tokens / pixels
#define NSEQn (NPx/64)       // 1536
#define LLn   4
#define DHd   24
#define EPSc  1e-5f

union HF { v16h v; float4 q[2]; };

static __device__ __forceinline__ float4 ldg4(const half_t* p) {
  return *(const float4*)p;
}
static __device__ __forceinline__ v8f wmma32(v16h a, v16h b, v8f c) {
  return __builtin_amdgcn_wmma_f32_16x16x32_f16(false, a, false, b, (short)0, c, false, false);
}
static __device__ __forceinline__ v8f vzero8() {
  v8f z;
  #pragma unroll
  for (int i = 0; i < 8; ++i) z[i] = 0.f;
  return z;
}
// A fragment: rows m0..m0+15 of row-major A (ld in halves), K-chunk [k0,k0+32)
static __device__ __forceinline__ v16h frag_a(const half_t* A, int ld, int m0, int k0) {
  int lane = threadIdx.x & 31;
  const half_t* p = A + (size_t)(m0 + (lane & 15)) * ld + k0 + ((lane & 16) ? 8 : 0);
  HF u; u.q[0] = ldg4(p); u.q[1] = ldg4(p + 16); return u.v;
}
// B fragment: rows n0..n0+15 of row-major W (N x K), K-chunk [k0,k0+32)
static __device__ __forceinline__ v16h frag_b(const half_t* W, int ld, int n0, int k0) {
  int lane = threadIdx.x & 31;
  const half_t* p = W + (size_t)(n0 + (lane & 15)) * ld + k0 + ((lane & 16) ? 8 : 0);
  HF u; u.q[0] = ldg4(p); u.q[1] = ldg4(p + 16); return u.v;
}
static __device__ __forceinline__ float gelu_f(float y) {
  return 0.5f * y * (1.f + erff(y * 0.70710678118f));
}

// ---------------- generic WMMA GEMM: Y[M x 96-slice] = A(MxK) @ W(96xK)^T + b ----
// EPI: 0 = f16 out, 1 = f32 out, 2 = gelu -> f16 out
template<int KDIM, int EPI>
__global__ __launch_bounds__(128)
void gemm6(const half_t* __restrict__ A, const half_t* __restrict__ W,
           const float* __restrict__ bias, half_t* __restrict__ Yh,
           float* __restrict__ Yf, int ldy, int col0)
{
  int wave = threadIdx.x >> 5, lane = threadIdx.x & 31, hi = lane >> 4;
  int m0 = blockIdx.x * 64 + wave * 16;
  v8f acc[6];
  #pragma unroll
  for (int nt = 0; nt < 6; ++nt) acc[nt] = vzero8();
  #pragma unroll
  for (int k0 = 0; k0 < KDIM; k0 += 32) {
    v16h a = frag_a(A, KDIM, m0, k0);
    #pragma unroll
    for (int nt = 0; nt < 6; ++nt)
      acc[nt] = wmma32(a, frag_b(W, KDIM, nt * 16, k0), acc[nt]);
  }
  int col = lane & 15;
  #pragma unroll
  for (int nt = 0; nt < 6; ++nt) {
    float bv = bias[nt * 16 + col];
    #pragma unroll
    for (int j = 0; j < 8; ++j) {
      float y = acc[nt][j] + bv;
      size_t o = (size_t)(m0 + hi * 8 + j) * ldy + col0 + nt * 16 + col;
      if (EPI == 0)      Yh[o] = (half_t)y;
      else if (EPI == 1) Yf[o] = y;
      else               Yh[o] = (half_t)gelu_f(y);
    }
  }
}

// ---------------- pointwise-conv GEMM for loc block: x += gelu(bn(pw(loc))) ------
__global__ __launch_bounds__(128)
void pw_gemm(const half_t* __restrict__ A, const half_t* __restrict__ W,
             const float* __restrict__ bs, const float* __restrict__ bb,
             const float* __restrict__ bm, const float* __restrict__ bvv,
             float* __restrict__ x, half_t* __restrict__ xh)
{
  int wave = threadIdx.x >> 5, lane = threadIdx.x & 31, hi = lane >> 4;
  int m0 = blockIdx.x * 64 + wave * 16;
  v8f acc[6];
  #pragma unroll
  for (int nt = 0; nt < 6; ++nt) acc[nt] = vzero8();
  #pragma unroll
  for (int k0 = 0; k0 < CCn; k0 += 32) {
    v16h a = frag_a(A, CCn, m0, k0);
    #pragma unroll
    for (int nt = 0; nt < 6; ++nt)
      acc[nt] = wmma32(a, frag_b(W, CCn, nt * 16, k0), acc[nt]);
  }
  int col = lane & 15;
  #pragma unroll
  for (int nt = 0; nt < 6; ++nt) {
    int c = nt * 16 + col;
    float inv = rsqrtf(bvv[c] + EPSc) * bs[c];
    float mu = bm[c], of = bb[c];
    #pragma unroll
    for (int j = 0; j < 8; ++j) {
      int p = m0 + hi * 8 + j;
      float y = gelu_f((acc[nt][j] - mu) * inv + of);
      int bq = p / TFP, rem = p % TFP;
      size_t xo = (size_t)(bq * CCn + c) * TFP + rem;
      float xn = x[xo] + y;
      x[xo] = xn;
      xh[(size_t)p * CCn + c] = (half_t)xn;
    }
  }
}

// ---------------- 3x3 conv as implicit WMMA GEMM (rf block) ----------------------
// EPI 1: y=leaky((bn1(conv)+beta)*(1+gamma)) -> t1h (f16 NHWC)
// EPI 2: x += bn2(conv); write x (f32 NCHW) + xh (f16 NHWC)
template<int EPI>
__global__ __launch_bounds__(128)
void rfconv(const half_t* __restrict__ Xh, const half_t* __restrict__ Wr,
            const float* __restrict__ bs, const float* __restrict__ bb,
            const float* __restrict__ bm, const float* __restrict__ bvv,
            const float* __restrict__ beta, const float* __restrict__ gamma,
            float* __restrict__ x, half_t* __restrict__ outh)
{
  int wave = threadIdx.x >> 5, lane = threadIdx.x & 31, hi = lane >> 4;
  int sid = blockIdx.x;
  int f0 = (sid & 3) << 6;
  int t  = (sid >> 2) % TTn;
  int bq = sid / (4 * TTn);
  int fa = f0 + wave * 16 + (lane & 15);   // this lane's A-row pixel (f coordinate)
  v8f acc[6];
  #pragma unroll
  for (int nt = 0; nt < 6; ++nt) acc[nt] = vzero8();
  #pragma unroll
  for (int dy = -1; dy <= 1; ++dy) {
    int tt = t + dy;
    bool tok = (tt >= 0) && (tt < TTn);
    #pragma unroll
    for (int dx = -1; dx <= 1; ++dx) {
      int ff = fa + dx;
      bool valid = tok && (ff >= 0) && (ff < FFn);
      const half_t* wb = Wr + ((dy + 1) * 3 + (dx + 1)) * CCn * CCn;
      #pragma unroll
      for (int k0 = 0; k0 < CCn; k0 += 32) {
        HF ua;
        if (valid) {
          const half_t* p = Xh + ((size_t)((bq * TTn + tt) * FFn) + ff) * CCn
                               + k0 + ((lane & 16) ? 8 : 0);
          ua.q[0] = ldg4(p); ua.q[1] = ldg4(p + 16);
        } else {
          ua.q[0] = make_float4(0.f, 0.f, 0.f, 0.f); ua.q[1] = ua.q[0];
        }
        #pragma unroll
        for (int nt = 0; nt < 6; ++nt)
          acc[nt] = wmma32(ua.v, frag_b(wb, CCn, nt * 16, k0), acc[nt]);
      }
    }
  }
  int col = lane & 15;
  #pragma unroll
  for (int nt = 0; nt < 6; ++nt) {
    int c = nt * 16 + col;
    float inv = rsqrtf(bvv[c] + EPSc) * bs[c];
    float mu = bm[c], of = bb[c];
    float be = (EPI == 1) ? beta[bq * CCn + c]  : 0.f;
    float ga = (EPI == 1) ? gamma[bq * CCn + c] : 0.f;
    #pragma unroll
    for (int j = 0; j < 8; ++j) {
      int f = f0 + wave * 16 + hi * 8 + j;
      int p = (bq * TTn + t) * FFn + f;
      float y = (acc[nt][j] - mu) * inv + of;
      if (EPI == 1) {
        y = (y + be) * (1.f + ga);
        y = (y >= 0.f) ? y : 0.01f * y;
        outh[(size_t)p * CCn + c] = (half_t)y;
      } else {
        size_t xo = (size_t)(bq * CCn + c) * TFP + t * FFn + f;
        float xn = x[xo] + y;
        x[xo] = xn;
        outh[(size_t)p * CCn + c] = (half_t)xn;
      }
    }
  }
}

// ---------------- attention: block = one 64-token sequence, wave = one head -----
__global__ __launch_bounds__(128)
void attn_kernel(const half_t* __restrict__ qkv, half_t* __restrict__ Oh)
{
  __shared__ __align__(16) half_t s_att[4][64 * 64];
  __shared__ __align__(16) half_t s_vt[4][32 * 64];
  int wave = threadIdx.x >> 5, lane = threadIdx.x & 31, hi = lane >> 4;
  int h = wave;
  int tokbase = blockIdx.x * 64;
  const half_t* base = qkv + (size_t)tokbase * 288;

  // stage V^T (rows 24..31 zero so N-tile 1 pads cleanly)
  for (int idx = lane; idx < 32 * 64; idx += 32) {
    int d = idx >> 6, kk = idx & 63;
    half_t v = (half_t)0.f;
    if (d < DHd) v = base[(size_t)kk * 288 + 192 + h * DHd + d];
    s_vt[wave][idx] = v;
  }

  // scores = Q @ K^T  (dh=24 zero-padded to K=32)
  v8f s[4][4];
  #pragma unroll
  for (int mt = 0; mt < 4; ++mt)
    for (int nt = 0; nt < 4; ++nt) s[mt][nt] = vzero8();

  HF ua[4];
  #pragma unroll
  for (int mt = 0; mt < 4; ++mt) {
    const half_t* qrow = base + (size_t)(mt * 16 + (lane & 15)) * 288 + h * DHd;
    if (!hi) { ua[mt].q[0] = ldg4(qrow);     ua[mt].q[1] = ldg4(qrow + 16); }
    else     { ua[mt].q[0] = ldg4(qrow + 8); ua[mt].q[1] = make_float4(0.f,0.f,0.f,0.f); }
  }
  #pragma unroll
  for (int nt = 0; nt < 4; ++nt) {
    const half_t* krow = base + (size_t)(nt * 16 + (lane & 15)) * 288 + 96 + h * DHd;
    HF ub;
    if (!hi) { ub.q[0] = ldg4(krow);      ub.q[1] = ldg4(krow + 16); }
    else     { ub.q[0] = ldg4(krow + 8);  ub.q[1] = make_float4(0.f,0.f,0.f,0.f); }
    #pragma unroll
    for (int mt = 0; mt < 4; ++mt)
      s[mt][nt] = wmma32(ua[mt].v, ub.v, s[mt][nt]);
  }

  // in-register softmax: row values of a C-tile live in one 16-lane half
  const float scale = 0.2041241452319315f;  // 1/sqrt(24)
  #pragma unroll
  for (int mt = 0; mt < 4; ++mt) {
    #pragma unroll
    for (int j = 0; j < 8; ++j) {
      float mx = -3.0e38f;
      #pragma unroll
      for (int nt = 0; nt < 4; ++nt) { s[mt][nt][j] *= scale; mx = fmaxf(mx, s[mt][nt][j]); }
      #pragma unroll
      for (int m = 8; m >= 1; m >>= 1) mx = fmaxf(mx, __shfl_xor(mx, m, 32));
      float e[4]; float sum = 0.f;
      #pragma unroll
      for (int nt = 0; nt < 4; ++nt) { e[nt] = __expf(s[mt][nt][j] - mx); sum += e[nt]; }
      #pragma unroll
      for (int m = 8; m >= 1; m >>= 1) sum += __shfl_xor(sum, m, 32);
      float r = 1.f / sum;
      int row = mt * 16 + hi * 8 + j;
      #pragma unroll
      for (int nt = 0; nt < 4; ++nt)
        s_att[wave][row * 64 + nt * 16 + (lane & 15)] = (half_t)(e[nt] * r);
    }
  }
  __syncthreads();

  // O = att @ V  (K=64, N=24 padded to 32)
  int col = lane & 15;
  #pragma unroll
  for (int mt = 0; mt < 4; ++mt) {
    v8f o0 = vzero8(), o1 = vzero8();
    #pragma unroll
    for (int k0 = 0; k0 < 64; k0 += 32) {
      v16h a = frag_a(&s_att[wave][0], 64, mt * 16, k0);
      o0 = wmma32(a, frag_b(&s_vt[wave][0], 64, 0,  k0), o0);
      o1 = wmma32(a, frag_b(&s_vt[wave][0], 64, 16, k0), o1);
    }
    #pragma unroll
    for (int j = 0; j < 8; ++j) {
      int tok = tokbase + mt * 16 + hi * 8 + j;
      Oh[(size_t)tok * 96 + h * DHd + col] = (half_t)o0[j];
      if (col < 8)
        Oh[(size_t)tok * 96 + h * DHd + 16 + col] = (half_t)o1[j];
    }
  }
}

// ---------------- LayerNorm(a+b) : one wave per token ---------------------------
__global__ __launch_bounds__(256)
void ln_add(const float* __restrict__ A, const float* __restrict__ B,
            const float* __restrict__ s, const float* __restrict__ b,
            float* __restrict__ O32, half_t* __restrict__ O16)
{
  int wave = threadIdx.x >> 5, lane = threadIdx.x & 31;
  size_t tok = (size_t)blockIdx.x * 8 + wave;
  const float* pa = A + tok * 96; const float* pb = B + tok * 96;
  float v[3]; float sum = 0.f;
  #pragma unroll
  for (int r = 0; r < 3; ++r) { int c = lane + 32 * r; v[r] = pa[c] + pb[c]; sum += v[r]; }
  #pragma unroll
  for (int m = 16; m >= 1; m >>= 1) sum += __shfl_xor(sum, m, 32);
  float mu = sum * (1.f / 96.f);
  float var = 0.f;
  #pragma unroll
  for (int r = 0; r < 3; ++r) { float d = v[r] - mu; var += d * d; }
  #pragma unroll
  for (int m = 16; m >= 1; m >>= 1) var += __shfl_xor(var, m, 32);
  float inv = rsqrtf(var * (1.f / 96.f) + EPSc);
  #pragma unroll
  for (int r = 0; r < 3; ++r) {
    int c = lane + 32 * r;
    float y = (v[r] - mu) * inv * s[c] + b[c];
    O32[tok * 96 + c] = y;
    O16[tok * 96 + c] = (half_t)y;
  }
}

// ---------------- LayerNorm(a+b) + scatter back to NCHW (undo roll) -------------
__global__ __launch_bounds__(256)
void ln2_scatter(const float* __restrict__ A, const float* __restrict__ B,
                 const float* __restrict__ s, const float* __restrict__ b,
                 float* __restrict__ xout, int mode, int shift)
{
  int wave = threadIdx.x >> 5, lane = threadIdx.x & 31;
  size_t tok = (size_t)blockIdx.x * 8 + wave;
  const float* pa = A + tok * 96; const float* pb = B + tok * 96;
  float v[3]; float sum = 0.f;
  #pragma unroll
  for (int r = 0; r < 3; ++r) { int c = lane + 32 * r; v[r] = pa[c] + pb[c]; sum += v[r]; }
  #pragma unroll
  for (int m = 16; m >= 1; m >>= 1) sum += __shfl_xor(sum, m, 32);
  float mu = sum * (1.f / 96.f);
  float var = 0.f;
  #pragma unroll
  for (int r = 0; r < 3; ++r) { float d = v[r] - mu; var += d * d; }
  #pragma unroll
  for (int m = 16; m >= 1; m >>= 1) var += __shfl_xor(var, m, 32);
  float inv = rsqrtf(var * (1.f / 96.f) + EPSc);
  int bq, t, f;
  if (mode == 0) { int fr = (int)(tok % FFn); int r2 = (int)(tok / FFn);
                   t = r2 % TTn; bq = r2 / TTn; f = (fr + shift) & (FFn - 1); }
  else           { int tr = (int)(tok % TTn); int r2 = (int)(tok / TTn);
                   f = r2 % FFn; bq = r2 / FFn; t = tr + shift; if (t >= TTn) t -= TTn; }
  #pragma unroll
  for (int r = 0; r < 3; ++r) {
    int c = lane + 32 * r;
    float y = (v[r] - mu) * inv * s[c] + b[c];
    xout[((size_t)(bq * CCn + c) * TTn + t) * FFn + f] = y;
  }
}

// ---------------- gather NCHW -> token-major (with roll shift), f32 + f16 -------
__global__ void gather_k(const float* __restrict__ x, float* __restrict__ X32,
                         half_t* __restrict__ X16, int mode, int shift)
{
  int idx = blockIdx.x * 256 + threadIdx.x;
  if (idx >= NPx * CCn) return;
  int c = idx % CCn; int tok = idx / CCn;
  int bq, t, f;
  if (mode == 0) { int fr = tok % FFn; int r2 = tok / FFn;
                   t = r2 % TTn; bq = r2 / TTn; f = (fr + shift) & (FFn - 1); }
  else           { int tr = tok % TTn; int r2 = tok / TTn;
                   f = r2 % FFn; bq = r2 / FFn; t = tr + shift; if (t >= TTn) t -= TTn; }
  float v = x[((size_t)(bq * CCn + c) * TTn + t) * FFn + f];
  X32[idx] = v; X16[idx] = (half_t)v;
}

// ---------------- x = bn(x + x_f + x_t); refresh NHWC f16 mirror ----------------
__global__ void combine_k(float* __restrict__ x, const float* __restrict__ xf,
                          const float* __restrict__ xt, const float* __restrict__ s,
                          const float* __restrict__ b, const float* __restrict__ m,
                          const float* __restrict__ vv, half_t* __restrict__ xh)
{
  int idx = blockIdx.x * 256 + threadIdx.x;
  if (idx >= BQn * CCn * TFP) return;
  int c = (idx / TFP) % CCn;
  float y = x[idx] + xf[idx] + xt[idx];
  y = (y - m[c]) * rsqrtf(vv[c] + EPSc) * s[c] + b[c];
  x[idx] = y;
  int bq = idx / (CCn * TFP); int rem = idx % TFP;
  xh[((size_t)bq * TFP + rem) * CCn + c] = (half_t)y;
}

// ---------------- depthwise 3x3 conv -> NHWC f16 --------------------------------
__global__ void dwconv_k(const float* __restrict__ x, const float* __restrict__ w,
                         half_t* __restrict__ loch)
{
  int idx = blockIdx.x * 256 + threadIdx.x;
  if (idx >= BQn * CCn * TFP) return;
  int f = idx % FFn; int t = (idx / FFn) % TTn;
  int c = (idx / TFP) % CCn; int bq = idx / (CCn * TFP);
  const float* wp = w + c * 9;
  float acc = 0.f;
  #pragma unroll
  for (int dy = -1; dy <= 1; ++dy) {
    int tt = t + dy; if (tt < 0 || tt >= TTn) continue;
    #pragma unroll
    for (int dx = -1; dx <= 1; ++dx) {
      int ff = f + dx; if (ff < 0 || ff >= FFn) continue;
      acc += x[((size_t)(bq * CCn + c) * TTn + tt) * FFn + ff] * wp[(dy + 1) * 3 + dx + 1];
    }
  }
  loch[((size_t)bq * TFP + t * FFn + f) * CCn + c] = (half_t)acc;
}

// ---------------- encoder conv (10->96 3x3) + bn + gelu -------------------------
__global__ void enc_k(const float* __restrict__ mr, const float* __restrict__ mi,
                      const float* __restrict__ er, const float* __restrict__ ei,
                      const float* __restrict__ w, const float* __restrict__ s,
                      const float* __restrict__ b, const float* __restrict__ m,
                      const float* __restrict__ vv,
                      float* __restrict__ x, half_t* __restrict__ xh)
{
  int idx = blockIdx.x * 256 + threadIdx.x;
  if (idx >= BQn * CCn * TFP) return;
  int f = idx % FFn; int t = (idx / FFn) % TTn;
  int o = (idx / TFP) % CCn; int bq = idx / (CCn * TFP);
  float acc = 0.f;
  for (int ic = 0; ic < 10; ++ic) {
    const float* src; size_t cho;
    if (ic < 4)       { src = mr; cho = (size_t)ic * TFP; }
    else if (ic == 4) { src = er; cho = (size_t)bq * TFP; }
    else if (ic < 9)  { src = mi; cho = (size_t)(ic - 5) * TFP; }
    else              { src = ei; cho = (size_t)bq * TFP; }
    #pragma unroll
    for (int dy = -1; dy <= 1; ++dy) {
      int tt = t + dy; if (tt < 0 || tt >= TTn) continue;
      #pragma unroll
      for (int dx = -1; dx <= 1; ++dx) {
        int ff = f + dx; if (ff < 0 || ff >= FFn) continue;
        acc += src[cho + (size_t)tt * FFn + ff] *
               w[((o * 10 + ic) * 3 + (dy + 1)) * 3 + (dx + 1)];
      }
    }
  }
  float y = (acc - m[o]) * rsqrtf(vv[o] + EPSc) * s[o] + b[o];
  y = gelu_f(y);
  x[idx] = y;
  xh[((size_t)bq * TFP + t * FFn + f) * CCn + o] = (half_t)y;
}

// ---------------- FiLM conditioning vectors -------------------------------------
__global__ void cond_k(const float* __restrict__ lbl,
                       const float* __restrict__ bw, const float* __restrict__ bbias,
                       const float* __restrict__ gw, const float* __restrict__ gbias,
                       float* __restrict__ beta, float* __restrict__ gamma)
{
  int tid = threadIdx.x; if (tid >= 192) return;
  int q = tid / 96, c = tid % 96;
  float sb = bbias[c], sg = gbias[c];
  for (int j = 0; j < 18; ++j) {
    float l = lbl[q * 18 + j];
    sb += l * bw[c * 18 + j];
    sg += l * gw[c * 18 + j];
  }
  beta[tid] = sb; gamma[tid] = sg;
}

// ---------------- head: tanh(1x1 conv) -> complex mask apply --------------------
__global__ void head_k(const float* __restrict__ x, const float* __restrict__ hw,
                       const float* __restrict__ hb, const float* __restrict__ mr,
                       const float* __restrict__ mi, float* __restrict__ out)
{
  int idx = blockIdx.x * 256 + threadIdx.x;
  if (idx >= BQn * TFP) return;
  int bq = idx / TFP; int rem = idx % TFP;
  float a0 = hb[0], a1 = hb[1];
  for (int c = 0; c < CCn; ++c) {
    float v = x[(size_t)(bq * CCn + c) * TFP + rem];
    a0 += hw[c] * v;
    a1 += hw[96 + c] * v;
  }
  float m0 = tanhf(a0), m1 = tanhf(a1);
  float rr = mr[rem], ri = mi[rem];            // mix channel 0
  out[((size_t)bq * 2 + 0) * TFP + rem] = m0 * rr - m1 * ri;
  out[((size_t)bq * 2 + 1) * TFP + rem] = m0 * ri + m1 * rr;
}

// ---------------- weight prep ---------------------------------------------------
__global__ void f2h_k(const float* __restrict__ in, half_t* __restrict__ out, int n) {
  int idx = blockIdx.x * 256 + threadIdx.x;
  if (idx < n) out[idx] = (half_t)in[idx];
}
// rf weights (L,Co,Ci,3,3) -> (L,9off,Co,Ci) f16 so B-rows are K-contiguous
__global__ void rfperm_k(const float* __restrict__ in, half_t* __restrict__ out) {
  int idx = blockIdx.x * 256 + threadIdx.x;
  if (idx >= LLn * 9 * CCn * CCn) return;
  int ci = idx % CCn; int co = (idx / CCn) % CCn;
  int off = (idx / (CCn * CCn)) % 9; int l = idx / (9 * CCn * CCn);
  out[idx] = (half_t)in[((size_t)(l * CCn + co) * CCn + ci) * 9 + off];
}

// =================================================================================
extern "C" void kernel_launch(void* const* d_in, const int* in_sizes, int n_in,
                              void* d_out, int out_size, void* d_ws, size_t ws_size,
                              hipStream_t stream)
{
  const float* mix_real  = (const float*)d_in[0];
  const float* mix_imag  = (const float*)d_in[1];
  const float* enr_real  = (const float*)d_in[2];
  const float* enr_imag  = (const float*)d_in[3];
  const float* label_vec = (const float*)d_in[4];
  const float* enc_w     = (const float*)d_in[5];
  const float* enc_bn_s  = (const float*)d_in[6];
  const float* enc_bn_b  = (const float*)d_in[7];
  const float* enc_bn_m  = (const float*)d_in[8];
  const float* enc_bn_v  = (const float*)d_in[9];
  const float* loc_dw    = (const float*)d_in[10];
  const float* loc_pw    = (const float*)d_in[11];
  const float* loc_bn_s  = (const float*)d_in[12];
  const float* loc_bn_b  = (const float*)d_in[13];
  const float* loc_bn_m  = (const float*)d_in[14];
  const float* loc_bn_v  = (const float*)d_in[15];
  const float* attn_in_w = (const float*)d_in[16];
  const float* attn_in_b = (const float*)d_in[17];
  const float* attn_out_w= (const float*)d_in[18];
  const float* attn_out_b= (const float*)d_in[19];
  const float* ln1_s     = (const float*)d_in[20];
  const float* ln1_b     = (const float*)d_in[21];
  const float* ln2_s     = (const float*)d_in[22];
  const float* ln2_b     = (const float*)d_in[23];
  const float* ff1_w     = (const float*)d_in[24];
  const float* ff1_b     = (const float*)d_in[25];
  const float* ff2_w     = (const float*)d_in[26];
  const float* ff2_b     = (const float*)d_in[27];
  const float* nbn_s     = (const float*)d_in[28];
  const float* nbn_b     = (const float*)d_in[29];
  const float* nbn_m     = (const float*)d_in[30];
  const float* nbn_v     = (const float*)d_in[31];
  const float* rf_c1     = (const float*)d_in[32];
  const float* rf_c2     = (const float*)d_in[33];
  const float* rf1_s = (const float*)d_in[34]; const float* rf1_b = (const float*)d_in[35];
  const float* rf1_m = (const float*)d_in[36]; const float* rf1_v = (const float*)d_in[37];
  const float* rf2_s = (const float*)d_in[38]; const float* rf2_b = (const float*)d_in[39];
  const float* rf2_m = (const float*)d_in[40]; const float* rf2_v = (const float*)d_in[41];
  const float* cbw = (const float*)d_in[42]; const float* cbb = (const float*)d_in[43];
  const float* cgw = (const float*)d_in[44]; const float* cgb = (const float*)d_in[45];
  const float* head_w = (const float*)d_in[46]; const float* head_b = (const float*)d_in[47];

  // ---- carve workspace ----
  char* p = (char*)d_ws;
  auto alloc = [&](size_t bytes) -> void* {
    void* r = (void*)p; p += (bytes + 255) & ~(size_t)255; return r;
  };
  const size_t NPC = (size_t)NPx * CCn;
  float*  x     = (float*) alloc(NPC * 4);
  half_t* xh    = (half_t*)alloc(NPC * 2);
  float*  x_f   = (float*) alloc(NPC * 4);
  float*  x_t   = (float*) alloc(NPC * 4);
  float*  X32   = (float*) alloc(NPC * 4);
  half_t* X16   = (half_t*)alloc(NPC * 2);
  half_t* qkvh  = (half_t*)alloc((size_t)NPx * 288 * 2);
  half_t* Oh    = (half_t*)alloc(NPC * 2);
  float*  O2    = (float*) alloc(NPC * 4);
  float*  x1_32 = (float*) alloc(NPC * 4);
  half_t* x1_16 = (half_t*)alloc(NPC * 2);
  half_t* ffh   = (half_t*)alloc((size_t)NPx * 192 * 2);
  float*  ff2f  = (float*) alloc(NPC * 4);
  half_t* loch  = (half_t*)alloc(NPC * 2);
  half_t* t1h   = (half_t*)alloc(NPC * 2);
  float*  beta  = (float*) alloc(192 * 4);
  float*  gamma = (float*) alloc(192 * 4);
  half_t* inwh  = (half_t*)alloc(2 * LLn * 288 * 96 * 2);
  half_t* outwh = (half_t*)alloc(2 * LLn * 96 * 96 * 2);
  half_t* ff1wh = (half_t*)alloc(2 * LLn * 192 * 96 * 2);
  half_t* ff2wh = (half_t*)alloc(2 * LLn * 96 * 192 * 2);
  half_t* pwh   = (half_t*)alloc(LLn * 96 * 96 * 2);
  half_t* rf1h  = (half_t*)alloc(LLn * 9 * 96 * 96 * 2);
  half_t* rf2h  = (half_t*)alloc(LLn * 9 * 96 * 96 * 2);

  auto gb = [](int n) { return (n + 255) / 256; };

  // ---- weight prep ----
  f2h_k<<<gb(2*LLn*288*96), 256, 0, stream>>>(attn_in_w, inwh, 2*LLn*288*96);
  f2h_k<<<gb(2*LLn*96*96),  256, 0, stream>>>(attn_out_w, outwh, 2*LLn*96*96);
  f2h_k<<<gb(2*LLn*192*96), 256, 0, stream>>>(ff1_w, ff1wh, 2*LLn*192*96);
  f2h_k<<<gb(2*LLn*96*192), 256, 0, stream>>>(ff2_w, ff2wh, 2*LLn*96*192);
  f2h_k<<<gb(LLn*96*96),    256, 0, stream>>>(loc_pw, pwh, LLn*96*96);
  rfperm_k<<<gb(LLn*9*96*96), 256, 0, stream>>>(rf_c1, rf1h);
  rfperm_k<<<gb(LLn*9*96*96), 256, 0, stream>>>(rf_c2, rf2h);
  cond_k<<<1, 192, 0, stream>>>(label_vec, cbw, cbb, cgw, cgb, beta, gamma);

  // ---- encoder conv ----
  enc_k<<<gb(BQn*CCn*TFP), 256, 0, stream>>>(mix_real, mix_imag, enr_real, enr_imag,
                                             enc_w, enc_bn_s, enc_bn_b, enc_bn_m, enc_bn_v,
                                             x, xh);

  for (int i = 0; i < LLn; ++i) {
    // local conv block
    dwconv_k<<<gb(BQn*CCn*TFP), 256, 0, stream>>>(x, loc_dw + i*CCn*9, loch);
    pw_gemm<<<NSEQn, 128, 0, stream>>>(loch, pwh + i*96*96,
                                       loc_bn_s + i*96, loc_bn_b + i*96,
                                       loc_bn_m + i*96, loc_bn_v + i*96, x, xh);
    int sh = (i & 1) ? 32 : 0;

    for (int axis = 0; axis < 2; ++axis) {    // 0 = freq, 1 = time
      int sl = axis * LLn + i;
      float* xdst = axis ? x_t : x_f;
      gather_k<<<gb((int)NPC), 256, 0, stream>>>(x, X32, X16, axis, sh);
      for (int part = 0; part < 3; ++part)
        gemm6<96,0><<<NSEQn, 128, 0, stream>>>(X16, inwh + (size_t)sl*288*96 + part*96*96,
                                               attn_in_b + sl*288 + part*96,
                                               qkvh, nullptr, 288, part*96);
      attn_kernel<<<NSEQn, 128, 0, stream>>>(qkvh, Oh);
      gemm6<96,1><<<NSEQn, 128, 0, stream>>>(Oh, outwh + (size_t)sl*96*96,
                                             attn_out_b + sl*96, nullptr, O2, 96, 0);
      ln_add<<<NPx/8, 256, 0, stream>>>(X32, O2, ln1_s + sl*96, ln1_b + sl*96, x1_32, x1_16);
      for (int part = 0; part < 2; ++part)
        gemm6<96,2><<<NSEQn, 128, 0, stream>>>(x1_16, ff1wh + ((size_t)sl*192 + part*96)*96,
                                               ff1_b + sl*192 + part*96,
                                               ffh, nullptr, 192, part*96);
      gemm6<192,1><<<NSEQn, 128, 0, stream>>>(ffh, ff2wh + (size_t)sl*96*192,
                                              ff2_b + sl*96, nullptr, ff2f, 96, 0);
      ln2_scatter<<<NPx/8, 256, 0, stream>>>(x1_32, ff2f, ln2_s + sl*96, ln2_b + sl*96,
                                             xdst, axis, sh);
    }
    combine_k<<<gb(BQn*CCn*TFP), 256, 0, stream>>>(x, x_f, x_t,
                                                   nbn_s + i*96, nbn_b + i*96,
                                                   nbn_m + i*96, nbn_v + i*96, xh);
    // refinement block
    rfconv<1><<<NSEQn, 128, 0, stream>>>(xh, rf1h + (size_t)i*9*96*96,
                                         rf1_s + i*96, rf1_b + i*96, rf1_m + i*96, rf1_v + i*96,
                                         beta, gamma, nullptr, t1h);
    rfconv<2><<<NSEQn, 128, 0, stream>>>(t1h, rf2h + (size_t)i*9*96*96,
                                         rf2_s + i*96, rf2_b + i*96, rf2_m + i*96, rf2_v + i*96,
                                         nullptr, nullptr, x, xh);
  }

  head_k<<<gb(BQn*TFP), 256, 0, stream>>>(x, head_w, head_b, mix_real, mix_imag,
                                          (float*)d_out);
}
